// Block_85383949845277
// MI455X (gfx1250) — matrix operand order repaired and checked by reading, not compiled, not run on hardware
//
#include <hip/hip_runtime.h>
#include <hip/hip_bf16.h>

// ---------------------------------------------------------------------------
// Transformer block for MI455X (gfx1250, wave32, WMMA + TDM).
// B=2, L=4096, D=512, H=8, hd=64. All GEMMs + attention einsums use
// v_wmma_f32_16x16x32_f16 (f16 operands, f32 accumulation). GEMM weight
// tiles are staged into LDS with the Tensor Data Mover (double-buffered,
// TENSORcnt-tracked) so weight traffic doesn't occupy wave VMEM issue slots.
// ---------------------------------------------------------------------------

typedef __attribute__((ext_vector_type(16))) _Float16 v16h;
typedef __attribute__((ext_vector_type(8)))  _Float16 v8h;
typedef __attribute__((ext_vector_type(8)))  float    v8f;

typedef __attribute__((ext_vector_type(4))) unsigned int tdm_u32x4;
typedef __attribute__((ext_vector_type(8))) int          tdm_i32x8;
typedef __attribute__((ext_vector_type(4))) int          tdm_i32x4;

#define WMMA_F16(a, b, c) \
  __builtin_amdgcn_wmma_f32_16x16x32_f16(false, (a), false, (b), (short)0, (c), false, false)

#if defined(__has_builtin)
#  if __has_builtin(__builtin_amdgcn_tensor_load_to_lds)
#    define USE_TDM 1
#  endif
#endif
#ifndef USE_TDM
#  define USE_TDM 0
#endif

// --- A fragment (16x32 f16, MxK) from row-major A[ld], per ISA 7.12.2 --------
// lane 0-15: M=lane, K = 0..7 (v0-3) and 16..23 (v4-7)
// lane 16-31: M=lane-16, K = 8..15 (v0-3) and 24..31 (v4-7)
static __device__ __forceinline__ v16h load_frag_a(const _Float16* A, int lda, int lane) {
  const _Float16* p = A + (size_t)(lane & 15) * lda + ((lane >> 4) << 3);
  v8h lo = *(const v8h*)(p);        // 16 contiguous bytes
  v8h hi = *(const v8h*)(p + 16);   // 16 contiguous bytes
  v16h r;
#pragma unroll
  for (int i = 0; i < 8; ++i) { r[i] = lo[i]; r[i + 8] = hi[i]; }
  return r;
}

// --- B fragment (32x16 f16, KxN) from row-major W (N rows x K cols)[ld] ------
// lane 0-15: N=lane, K = 0..7 and 16..23 ; lane 16-31: N=lane-16, K = 8..15 and 24..31
static __device__ __forceinline__ v16h load_frag_b(const _Float16* W, int ldw, int lane) {
  const _Float16* p = W + (size_t)(lane & 15) * ldw + ((lane >> 4) << 3);
  v8h lo = *(const v8h*)(p);
  v8h hi = *(const v8h*)(p + 16);
  v16h r;
#pragma unroll
  for (int i = 0; i < 8; ++i) { r[i] = lo[i]; r[i + 8] = hi[i]; }
  return r;
}

#if USE_TDM
// --- Tensor Data Mover: 2-D tile (64 rows x tile_k f16) global -> LDS --------
// D# built per CDNA5 ISA 8.3/8.4: group0 = {count, lds_addr, global_addr, type},
// group1 = {data_size, tensor_dim0/1, tile_dim0/1, tensor_dim0_stride}.
static __device__ __forceinline__ void tdm_load_tile(unsigned lds_byte_off,
                                                     const _Float16* gptr,
                                                     unsigned k_elems,   // tensor_dim0 & row stride
                                                     unsigned n_rows,    // tensor_dim1
                                                     unsigned tile_k) {  // tile_dim0
  const unsigned long long ga = (unsigned long long)(uintptr_t)gptr;
  tdm_u32x4 g0;
  g0[0] = 1u;                                                 // count = 1 (valid D#)
  g0[1] = lds_byte_off;                                       // lds_addr
  g0[2] = (unsigned)ga;                                       // global_addr[31:0]
  g0[3] = (unsigned)((ga >> 32) & 0x01FFFFFFu) | (2u << 30);  // addr[56:32] | type=2
  tdm_i32x8 g1;
  g1[0] = (int)(1u << 16);                                    // data_size = 2 bytes
  g1[1] = (int)((k_elems & 0xFFFFu) << 16);                   // tensor_dim0[15:0]
  g1[2] = (int)((k_elems >> 16) | ((n_rows & 0xFFFFu) << 16));// dim0[31:16]|dim1[15:0]
  g1[3] = (int)((n_rows >> 16) | (tile_k << 16));             // dim1[31:16]|tile_dim0
  g1[4] = 64;                                                 // tile_dim1=64, tile_dim2=0
  g1[5] = (int)k_elems;                                       // tensor_dim0_stride[31:0]
  g1[6] = 0;
  g1[7] = 0;
  tdm_i32x4 g2 = (tdm_i32x4)0;
  tdm_i32x4 g3 = (tdm_i32x4)0;
#if defined(__clang_major__) && __clang_major__ >= 23
  tdm_i32x8 g4 = (tdm_i32x8)0;
  __builtin_amdgcn_tensor_load_to_lds(g0, g1, g2, g3, g4, 0);
#else
  __builtin_amdgcn_tensor_load_to_lds(g0, g1, g2, g3, 0);
#endif
}
#endif  // USE_TDM

// ---------------------------------------------------------------------------
// f32 -> f16 elementwise (weight down-conversion)
// ---------------------------------------------------------------------------
__global__ void f32_to_f16_kernel(const float* __restrict__ in,
                                  _Float16* __restrict__ out, int n) {
  int i = blockIdx.x * blockDim.x + threadIdx.x;
  if (i < n) out[i] = (_Float16)in[i];
}

// ---------------------------------------------------------------------------
// LayerNorm over D=512, one wave per row, wave32 shfl reductions, f16 output
// ---------------------------------------------------------------------------
__global__ __launch_bounds__(256)
void layernorm_f16_kernel(const float* __restrict__ x, const float* __restrict__ w,
                          _Float16* __restrict__ out) {
  const int lane = threadIdx.x & 31;
  const int wave = threadIdx.x >> 5;
  const int row  = blockIdx.x * 8 + wave;
  const float* xr = x + (size_t)row * 512 + lane * 16;
  float v[16];
  float s = 0.f, ss = 0.f;
#pragma unroll
  for (int i = 0; i < 4; ++i) {
    float4 t = *(const float4*)(xr + i * 4);
    v[i*4+0] = t.x; v[i*4+1] = t.y; v[i*4+2] = t.z; v[i*4+3] = t.w;
    s  += t.x + t.y + t.z + t.w;
    ss += t.x*t.x + t.y*t.y + t.z*t.z + t.w*t.w;
  }
#pragma unroll
  for (int off = 16; off; off >>= 1) {
    s  += __shfl_xor(s,  off, 32);
    ss += __shfl_xor(ss, off, 32);
  }
  const float mu  = s * (1.f / 512.f);
  const float inv = rsqrtf(ss * (1.f / 512.f) - mu * mu + 1e-6f);
  const float* wr = w + lane * 16;
  v8h o0, o1;
#pragma unroll
  for (int i = 0; i < 8; ++i) o0[i] = (_Float16)((v[i]     - mu) * inv * wr[i]);
#pragma unroll
  for (int i = 0; i < 8; ++i) o1[i] = (_Float16)((v[i + 8] - mu) * inv * wr[i + 8]);
  _Float16* orow = out + (size_t)row * 512 + lane * 16;
  *(v8h*)(orow)     = o0;
  *(v8h*)(orow + 8) = o1;
}

// ---------------------------------------------------------------------------
// Generic C = A @ W^T GEMM on WMMA. A: MxK f16 row-major. W: NxK f16 row-major.
// ACT==1 -> exact-erf GELU. RES -> add f32 residual (same MxN layout).
// 64x64 tile per 128-thread block, 32x32 (2x2 WMMA tiles) per wave.
// W tile double-buffered through LDS via TDM (tensor_load_to_lds), pipelined
// on TENSORcnt; A streams directly from global (each A row used by one block
// row only per n-panel, served from L2).
// ---------------------------------------------------------------------------
template <int ACT, bool RES>
__global__ __launch_bounds__(128)
void gemm_bt_kernel(const _Float16* __restrict__ A, const _Float16* __restrict__ W,
                    const float* __restrict__ res, float* __restrict__ Cf,
                    _Float16* __restrict__ Ch, int M, int N, int K) {
  constexpr int KC = 128;                     // K-chunk staged per TDM transfer
  extern __shared__ _Float16 smem[];          // 2 x (64 x KC) f16 = 32 KB
  const int lane = threadIdx.x & 31;
  const int wave = threadIdx.x >> 5;
  const int m0   = blockIdx.y * 64 + (wave >> 1) * 32;
  const int nblk = blockIdx.x * 64;
  const int nloc = (wave & 1) * 32;
  v8f acc00 = {}, acc01 = {}, acc10 = {}, acc11 = {};
  const _Float16* Am0  = A + (size_t)m0 * K;
  const _Float16* Am1  = A + (size_t)(m0 + 16) * K;
  const _Float16* Wblk = W + (size_t)nblk * K;

#if USE_TDM
  if (wave == 0) tdm_load_tile(0, Wblk, K, N, KC);   // prime buffer 0
  int cur = 0;
  for (int kc = 0; kc < K; kc += KC) {
    if (wave == 0) {
      if (kc + KC < K) {
        tdm_load_tile((unsigned)((cur ^ 1) * (64 * KC * 2)), Wblk + (kc + KC), K, N, KC);
        __builtin_amdgcn_s_wait_tensorcnt(1);  // current buffer landed
      } else {
        __builtin_amdgcn_s_wait_tensorcnt(0);
      }
    }
    __syncthreads();                            // publish LDS tile to all waves
    const _Float16* Wl = smem + (size_t)cur * (64 * KC);
#pragma unroll
    for (int k0 = 0; k0 < KC; k0 += 32) {
      v16h a0 = load_frag_a(Am0 + kc + k0, K, lane);
      v16h a1 = load_frag_a(Am1 + kc + k0, K, lane);
      v16h b0 = load_frag_b(Wl + (size_t)nloc * KC + k0, KC, lane);
      v16h b1 = load_frag_b(Wl + (size_t)(nloc + 16) * KC + k0, KC, lane);
      acc00 = WMMA_F16(a0, b0, acc00);
      acc01 = WMMA_F16(a0, b1, acc01);
      acc10 = WMMA_F16(a1, b0, acc10);
      acc11 = WMMA_F16(a1, b1, acc11);
      if (k0 == 0) __builtin_prefetch(Am0 + kc + KC, 0, 0);  // next A chunk
    }
    __syncthreads();                            // done reading before refill
    cur ^= 1;
  }
#else
  const _Float16* Wn0 = W + (size_t)(nblk + nloc) * K;
  const _Float16* Wn1 = W + (size_t)(nblk + nloc + 16) * K;
  for (int k0 = 0; k0 < K; k0 += 32) {
    v16h a0 = load_frag_a(Am0 + k0, K, lane);
    v16h a1 = load_frag_a(Am1 + k0, K, lane);
    v16h b0 = load_frag_b(Wn0 + k0, K, lane);
    v16h b1 = load_frag_b(Wn1 + k0, K, lane);
    acc00 = WMMA_F16(a0, b0, acc00);
    acc01 = WMMA_F16(a0, b1, acc01);
    acc10 = WMMA_F16(a1, b0, acc10);
    acc11 = WMMA_F16(a1, b1, acc11);
  }
#endif

  const int n0 = nblk + nloc;
  const int cl = lane & 15;
  const int rb = (lane >> 4) * 8;
#pragma unroll
  for (int mt = 0; mt < 2; ++mt) {
#pragma unroll
    for (int nt = 0; nt < 2; ++nt) {
      v8f c = (mt == 0) ? (nt == 0 ? acc00 : acc01) : (nt == 0 ? acc10 : acc11);
#pragma unroll
      for (int j = 0; j < 8; ++j) {
        const int row = m0 + mt * 16 + rb + j;
        const int col = n0 + nt * 16 + cl;
        float v = c[j];
        if (ACT == 1) v = 0.5f * v * (1.0f + erff(v * 0.70710678118f));
        if (RES) v += res[(size_t)row * N + col];
        if (Cf) Cf[(size_t)row * N + col] = v;
        if (Ch) Ch[(size_t)row * N + col] = (_Float16)v;
      }
    }
  }
}

// ---------------------------------------------------------------------------
// V transpose: qkv (8192 x 1536, cols 1024..1535 = V) -> vT[(b,h), d, l]
// so that P@V's B fragment reads contiguous keys per lane.
// ---------------------------------------------------------------------------
__global__ void transpose_v_kernel(const _Float16* __restrict__ qkv,
                                   _Float16* __restrict__ vT) {
  const int idx = blockIdx.x * blockDim.x + threadIdx.x;   // 16*64*4096 total
  const int l  = idx & 4095;
  const int d  = (idx >> 12) & 63;
  const int bh = idx >> 18;
  const int b = bh >> 3, h = bh & 7;
  vT[idx] = qkv[((size_t)(b * 4096 + l)) * 1536 + 1024 + h * 64 + d];
}

// ---------------------------------------------------------------------------
// Causal flash attention. One wave = one 16-query tile of one (b,h).
// S = Q K^T via 4 WMMAs per 32-key block; online softmax with half-wave
// shfl row reductions (C-layout rows live in VGPR index, cols in lanes);
// P re-laid-out through LDS into an A fragment; O += P V via 4 WMMAs.
// ---------------------------------------------------------------------------
__global__ __launch_bounds__(128)
void attention_kernel(const _Float16* __restrict__ qkv,
                      const _Float16* __restrict__ vT,
                      _Float16* __restrict__ o) {
  __shared__ _Float16 lds_p[4][16 * 32];   // 1 KB per wave
  const int lane = threadIdx.x & 31;
  const int wave = threadIdx.x >> 5;
  const int tile = blockIdx.x * 4 + wave;  // 0..4095
  const int qt = tile & 255;               // L/16 = 256 query tiles
  const int bh = tile >> 8;                // 0..15
  const int b = bh >> 3, h = bh & 7;
  const int q0 = qt * 16;

  const _Float16* qbase = qkv + ((size_t)(b * 4096 + q0)) * 1536 + h * 64;
  const _Float16* kbase = qkv + ((size_t)(b * 4096)) * 1536 + 512 + h * 64;
  const _Float16* vbase = vT + (size_t)bh * 64 * 4096;

  const v16h qa0 = load_frag_a(qbase,      1536, lane);  // K-dims 0..31
  const v16h qa1 = load_frag_a(qbase + 32, 1536, lane);  // K-dims 32..63

  v8f acc0 = {}, acc1 = {}, acc2 = {}, acc3 = {};
  float rm[8], rl[8];
#pragma unroll
  for (int j = 0; j < 8; ++j) { rm[j] = -1e30f; rl[j] = 0.f; }

  const int cl = lane & 15;
  const int rb = (lane >> 4) * 8;
  _Float16* lp = lds_p[wave];
  const float scale = 0.125f;  // 1/sqrt(64)

  for (int kb = 0; kb < q0 + 16; kb += 32) {
    // ---- S = Q K^T for keys [kb, kb+32) --------------------------------
    const _Float16* kr = kbase + (size_t)kb * 1536;
    v16h b00 = load_frag_b(kr,                  1536, lane);
    v16h b01 = load_frag_b(kr + 32,             1536, lane);
    v16h b10 = load_frag_b(kr + 16 * 1536,      1536, lane);
    v16h b11 = load_frag_b(kr + 16 * 1536 + 32, 1536, lane);
    v8f s0 = {}, s1 = {};
    s0 = WMMA_F16(qa0, b00, s0);
    s0 = WMMA_F16(qa1, b01, s0);
    s1 = WMMA_F16(qa0, b10, s1);
    s1 = WMMA_F16(qa1, b11, s1);

    // ---- online softmax (per-row; rows = VGPR idx, cols = lanes) -------
#pragma unroll
    for (int j = 0; j < 8; ++j) {
      const int row = q0 + rb + j;
      const int c0 = kb + cl, c1 = kb + 16 + cl;
      float v0 = (c0 <= row) ? s0[j] * scale : -1e30f;   // causal mask
      float v1 = (c1 <= row) ? s1[j] * scale : -1e30f;
      float mc = fmaxf(v0, v1);
#pragma unroll
      for (int off = 8; off; off >>= 1) mc = fmaxf(mc, __shfl_xor(mc, off, 16));
      const float mnew = fmaxf(rm[j], mc);
      const float ef = __expf(rm[j] - mnew);
      rm[j] = mnew;
      const float p0 = __expf(v0 - mnew);
      const float p1 = __expf(v1 - mnew);
      float rs = p0 + p1;
#pragma unroll
      for (int off = 8; off; off >>= 1) rs += __shfl_xor(rs, off, 16);
      rl[j] = rl[j] * ef + rs;
      acc0[j] *= ef; acc1[j] *= ef; acc2[j] *= ef; acc3[j] *= ef;
      lp[(rb + j) * 32 + cl]      = (_Float16)p0;   // ds_store: re-layout P
      lp[(rb + j) * 32 + cl + 16] = (_Float16)p1;
    }
    asm volatile("s_wait_dscnt 0" ::: "memory");      // LDS writes visible

    // ---- O += P V ------------------------------------------------------
    const v16h pa = load_frag_a(lp, 32, lane);        // ds_load_b128 x2
    acc0 = WMMA_F16(pa, load_frag_b(vbase + (size_t)0 * 16 * 4096 + kb, 4096, lane), acc0);
    acc1 = WMMA_F16(pa, load_frag_b(vbase + (size_t)1 * 16 * 4096 + kb, 4096, lane), acc1);
    acc2 = WMMA_F16(pa, load_frag_b(vbase + (size_t)2 * 16 * 4096 + kb, 4096, lane), acc2);
    acc3 = WMMA_F16(pa, load_frag_b(vbase + (size_t)3 * 16 * 4096 + kb, 4096, lane), acc3);
  }

  // ---- normalize + store f16 o rows ------------------------------------
  _Float16* orow = o + ((size_t)(b * 4096 + q0 + rb)) * 512 + h * 64 + cl;
#pragma unroll
  for (int j = 0; j < 8; ++j) {
    const float inv = 1.0f / rl[j];
    orow[(size_t)j * 512 + 0]  = (_Float16)(acc0[j] * inv);
    orow[(size_t)j * 512 + 16] = (_Float16)(acc1[j] * inv);
    orow[(size_t)j * 512 + 32] = (_Float16)(acc2[j] * inv);
    orow[(size_t)j * 512 + 48] = (_Float16)(acc3[j] * inv);
  }
}

// ---------------------------------------------------------------------------
// Host-side orchestration
// ---------------------------------------------------------------------------
extern "C" void kernel_launch(void* const* d_in, const int* in_sizes, int n_in,
                              void* d_out, int out_size, void* d_ws, size_t ws_size,
                              hipStream_t stream) {
  const float* x      = (const float*)d_in[0];
  // d_in[1] = mask (bool, causal) — applied analytically in attention_kernel.
  const float* ln1_w  = (const float*)d_in[2];
  const float* w_qkv  = (const float*)d_in[3];
  const float* w_out  = (const float*)d_in[4];
  const float* ln2_w  = (const float*)d_in[5];
  const float* w_mlp1 = (const float*)d_in[6];
  const float* w_mlp2 = (const float*)d_in[7];
  float* out = (float*)d_out;

  constexpr int M = 2 * 4096;   // token rows
  constexpr int D = 512;
  constexpr size_t GEMM_LDS = 2 * 64 * 128 * sizeof(_Float16);  // 32 KB

  char* ws = (char*)d_ws;
  size_t off = 0;
  auto alloc = [&](size_t bytes) -> char* {
    char* p = ws + off;
    off += (bytes + 255) & ~(size_t)255;
    return p;
  };
  _Float16* wqkv_h  = (_Float16*)alloc((size_t)1536 * 512 * 2);
  _Float16* wout_h  = (_Float16*)alloc((size_t)512  * 512 * 2);
  _Float16* wmlp1_h = (_Float16*)alloc((size_t)2048 * 512 * 2);
  _Float16* wmlp2_h = (_Float16*)alloc((size_t)512  * 2048 * 2);
  _Float16* n1      = (_Float16*)alloc((size_t)M * D * 2);
  _Float16* qkv_h   = (_Float16*)alloc((size_t)M * 1536 * 2);
  _Float16* vT      = (_Float16*)alloc((size_t)M * D * 2);
  _Float16* o_h     = (_Float16*)alloc((size_t)M * D * 2);
  float*    x_a     = (float*)   alloc((size_t)M * D * 4);
  _Float16* n2      = (_Float16*)alloc((size_t)M * D * 2);
  _Float16* h_h     = (_Float16*)alloc((size_t)M * 2048 * 2);

  // 1) weights -> f16
  f32_to_f16_kernel<<<(1536 * 512 + 255) / 256, 256, 0, stream>>>(w_qkv,  wqkv_h,  1536 * 512);
  f32_to_f16_kernel<<<(512  * 512 + 255) / 256, 256, 0, stream>>>(w_out,  wout_h,  512  * 512);
  f32_to_f16_kernel<<<(2048 * 512 + 255) / 256, 256, 0, stream>>>(w_mlp1, wmlp1_h, 2048 * 512);
  f32_to_f16_kernel<<<(512 * 2048 + 255) / 256, 256, 0, stream>>>(w_mlp2, wmlp2_h, 512 * 2048);

  // 2) n1 = LN(x)
  layernorm_f16_kernel<<<M / 8, 256, 0, stream>>>(x, ln1_w, n1);

  // 3) qkv = n1 @ w_qkv^T
  gemm_bt_kernel<0, false><<<dim3(1536 / 64, M / 64), 128, GEMM_LDS, stream>>>(
      n1, wqkv_h, nullptr, nullptr, qkv_h, M, 1536, 512);

  // 4) vT layout for P@V
  transpose_v_kernel<<<(M * D) / 256, 256, 0, stream>>>(qkv_h, vT);

  // 5) causal flash attention -> o (f16)
  attention_kernel<<<(2 * 8 * 256) / 4, 128, 0, stream>>>(qkv_h, vT, o_h);

  // 6) x_a = x + o @ w_out^T
  gemm_bt_kernel<0, true><<<dim3(512 / 64, M / 64), 128, GEMM_LDS, stream>>>(
      o_h, wout_h, x, x_a, nullptr, M, 512, 512);

  // 7) n2 = LN(x_a)
  layernorm_f16_kernel<<<M / 8, 256, 0, stream>>>(x_a, ln2_w, n2);

  // 8) h = gelu(n2 @ w_mlp1^T)
  gemm_bt_kernel<1, false><<<dim3(2048 / 64, M / 64), 128, GEMM_LDS, stream>>>(
      n2, wmlp1_h, nullptr, nullptr, h_h, M, 2048, 512);

  // 9) out = x_a + h @ w_mlp2^T
  gemm_bt_kernel<0, true><<<dim3(512 / 64, M / 64), 128, GEMM_LDS, stream>>>(
      h_h, wmlp2_h, x_a, out, nullptr, M, 512, 2048);
}
